// DAPSO_54305566491176
// MI455X (gfx1250) — matrix operand order repaired and checked by reading, not compiled, not run on hardware
//
#include <hip/hip_runtime.h>

typedef __bf16 v16bf __attribute__((ext_vector_type(16)));
typedef __bf16 v8bf  __attribute__((ext_vector_type(8)));
typedef float  v8f   __attribute__((ext_vector_type(8)));
typedef unsigned short u16;

union U16F { v16bf v; v8bf h[2]; };

__device__ __forceinline__ float gelu_f(float v) {
    return 0.5f * v * (1.0f + erff(v * 0.70710678118654752f));
}
__device__ __forceinline__ float softplus_f(float x) {
    return (x > 20.0f) ? x : log1pf(expf(x));
}
__device__ __forceinline__ u16 f2bf(float f) {
    unsigned int u = __float_as_uint(f);
    unsigned int r = (u + 0x7FFFu + ((u >> 16) & 1u)) >> 16;
    return (u16)r;
}
__device__ __forceinline__ float bf2f(u16 h) {
    return __uint_as_float(((unsigned int)h) << 16);
}

// ---------------------------------------------------------------------------
// Workspace layout (float offsets unless noted)
// ---------------------------------------------------------------------------
#define OFF_GH 0u            // 129*128
#define OFF_GW 16512u
#define OFF_KH 33024u        // 128*256
#define OFF_KW 65792u
#define OFF_P  98560u        // 4*256 pooled means
#define OFF_SC 99584u        // 256 bn scale
#define OFF_BI 99840u        // 256 bn bias
#define OFF_MH 100352u       // 128*256*256
#define OFF_MW_F (OFF_MH + 8388608u)
#define AB_BYTE_OFF  ((size_t)(OFF_MW_F + 8388608u) * 4u)   // 67,510,272
#define YBF_BYTE_OFF (AB_BYTE_OFF + 524288u)                // 68,034,560

// ---------------------------------------------------------------------------
// Kernel 1: tiny MLPs -> gate tables Gh, Gw (129 x 128 each)
// ---------------------------------------------------------------------------
__device__ void mlp1(float g,
                     const float* __restrict__ w1, const float* __restrict__ b1,
                     const float* __restrict__ w2, const float* __restrict__ b2,
                     const float* __restrict__ w3, const float* __restrict__ b3,
                     float* out8) {
    float h1[64], h2[64];
    for (int i = 0; i < 64; ++i) h1[i] = gelu_f(w1[i] * g + b1[i]);
    for (int i = 0; i < 64; ++i) {
        float s = b2[i];
        for (int k = 0; k < 64; ++k) s += w2[i * 64 + k] * h1[k];
        h2[i] = gelu_f(s);
    }
    for (int r = 0; r < 8; ++r) {
        float s = b3[r];
        for (int k = 0; k < 64; ++k) s += w3[r * 64 + k] * h2[k];
        out8[r] = s;
    }
}

__global__ void gates_kernel(
    const float* ah_w1, const float* ah_b1, const float* ah_w2, const float* ah_b2,
    const float* ah_w3, const float* ah_b3,
    const float* bc1_w1, const float* bc1_b1, const float* bc1_w2, const float* bc1_b2,
    const float* bc1_w3, const float* bc1_b3,
    const float* aw_w1, const float* aw_b1, const float* aw_w2, const float* aw_b2,
    const float* aw_w3, const float* aw_b3,
    const float* bc2_w1, const float* bc2_b1, const float* bc2_w2, const float* bc2_b2,
    const float* bc2_w3, const float* bc2_b3,
    float* __restrict__ Gh, float* __restrict__ Gw) {
    __shared__ float lAh[129 * 8], lAw[129 * 8], lBh[128 * 8], lBw[128 * 8];
    int t = threadIdx.x;  // 1024 threads
    float o8[8];
    if (t < 129) {
        float om = (float)t * (1.0f / 128.0f) - 1.0f;
        mlp1(om, ah_w1, ah_b1, ah_w2, ah_b2, ah_w3, ah_b3, o8);
        for (int r = 0; r < 8; ++r) lAh[t * 8 + r] = o8[r];
    } else if (t >= 256 && t < 385) {
        int h = t - 256;
        float om = (float)h * (1.0f / 128.0f) - 1.0f;
        mlp1(om, aw_w1, aw_b1, aw_w2, aw_b2, aw_w3, aw_b3, o8);
        for (int r = 0; r < 8; ++r) lAw[h * 8 + r] = o8[r];
    } else if (t >= 512 && t < 640) {
        int c = t - 512;
        float lam = -1.0f + 2.0f * (float)c / 127.0f;
        mlp1(lam, bc1_w1, bc1_b1, bc1_w2, bc1_b2, bc1_w3, bc1_b3, o8);
        for (int r = 0; r < 8; ++r) lBh[c * 8 + r] = o8[r];
    } else if (t >= 768 && t < 896) {
        int c = t - 768;
        float lam = -1.0f + 2.0f * (float)c / 127.0f;
        mlp1(lam, bc2_w1, bc2_b1, bc2_w2, bc2_b2, bc2_w3, bc2_b3, o8);
        for (int r = 0; r < 8; ++r) lBw[c * 8 + r] = o8[r];
    }
    __syncthreads();
    const float invR = 0.35355339059327373f;  // 1/sqrt(8)
    for (int idx = t; idx < 129 * 128; idx += 1024) {
        int h = idx >> 7, c = idx & 127;
        float s = 0.f, s2 = 0.f;
        for (int r = 0; r < 8; ++r) {
            s  += lAh[h * 8 + r] * lBh[c * 8 + r];
            s2 += lAw[h * 8 + r] * lBw[c * 8 + r];
        }
        Gh[idx] = softplus_f(s) * invR;
        Gw[idx] = softplus_f(s2) * invR;
    }
}

// ---------------------------------------------------------------------------
// Kernel 2: circulant kernel K_c[d] from gate column c
// ---------------------------------------------------------------------------
__global__ void kbuild_kernel(const float* __restrict__ Gh, const float* __restrict__ Gw,
                              float* __restrict__ Kh, float* __restrict__ Kw) {
    int c = blockIdx.x;
    const float* G = blockIdx.y ? Gw : Gh;
    float* K = blockIdx.y ? Kw : Kh;
    __shared__ float gl[129];
    if (threadIdx.x < 129) gl[threadIdx.x] = G[threadIdx.x * 128 + c];
    __syncthreads();
    int d = threadIdx.x;  // 256
    float s = gl[0];
    const float PI_128 = 0.02454369260617026f;  // pi/128
    for (int k = 1; k < 128; ++k) {
        int idx = (k * d) & 255;
        s += 2.0f * gl[k] * cosf((float)idx * PI_128);
    }
    s += gl[128] * ((d & 1) ? -1.0f : 1.0f);
    K[c * 256 + d] = s * (1.0f / 256.0f);
}

// ---------------------------------------------------------------------------
// Kernel 3: materialize M_c[n,m] = K_c[(n-m) & 255]
// ---------------------------------------------------------------------------
__global__ void mbuild_kernel(const float* __restrict__ Kh, const float* __restrict__ Kw,
                              float* __restrict__ Mh, float* __restrict__ Mw) {
    int c = blockIdx.x;
    const float* K = blockIdx.y ? Kw : Kh;
    float* M = (blockIdx.y ? Mw : Mh) + (size_t)c * 65536;
    __shared__ float kl[256];
    kl[threadIdx.x] = K[c * 256 + threadIdx.x];
    __syncthreads();
    int m = threadIdx.x;
    for (int n = 0; n < 256; ++n)
        M[n * 256 + m] = kl[(n - m) & 255];
}

// ---------------------------------------------------------------------------
// Kernel 4: per-channel spectral GEMM (WMMA bf16, f32 accumulate)
//   m_on_left=1: Y = M_c @ X   (HC branch, channels 0..127)
//   m_on_left=0: Y = X @ M_c   (WC branch, channels 128..255; M symmetric)
// Double-buffered LDS with register-staged prefetch: global loads for tile
// i+1 issue before the WMMAs of tile i, so HBM latency hides under compute.
// Writes y as bf16. grid=(2,2,512) block=256.
// ---------------------------------------------------------------------------
__global__ void spectral_gemm(const float* __restrict__ Mbase,
                              const float* __restrict__ xbase,
                              u16* __restrict__ ybase, int m_on_left) {
    int c = blockIdx.z >> 2;
    int b = blockIdx.z & 3;
    int chan = m_on_left ? c : (128 + c);
    size_t slab = ((size_t)(b * 256 + chan)) * 65536u;
    const float* Mc = Mbase + (size_t)c * 65536u;
    const float* Xc = xbase + slab;
    const float* A = m_on_left ? Mc : Xc;
    const float* B = m_on_left ? Xc : Mc;
    u16* C = ybase + slab;

    __shared__ u16 ldsA[2][128 * 32];   // (m, k) row-major
    __shared__ u16 ldsBt[2][128 * 32];  // (n, k) row-major (transposed stage)

    int tid = threadIdx.x;
    int lane = tid & 31;
    int wave = tid >> 5;
    int row0 = blockIdx.x * 128;
    int col0 = blockIdx.y * 128;
    int kg = lane >> 4;
    int l16 = lane & 15;

    // per-thread staging coordinates (fixed across k-steps)
    int ar[4], ac[4], bk[4], bn_[4];
#pragma unroll
    for (int it = 0; it < 4; ++it) {
        int idx = tid + it * 256;            // 1024 float4 chunks per operand
        ar[it] = idx >> 3;  ac[it] = (idx & 7) * 4;    // A: 8 chunks/row
        bk[it] = idx >> 5;  bn_[it] = (idx & 31) * 4;  // B: 32 chunks/k-row
    }

    v8f acc[8];
#pragma unroll
    for (int j = 0; j < 8; ++j) acc[j] = (v8f)(0.0f);

    float4 ra[4], rb[4];
    // prologue: load + stage tile 0 into buffer 0
#pragma unroll
    for (int it = 0; it < 4; ++it) {
        ra[it] = *(const float4*)(A + (size_t)(row0 + ar[it]) * 256 + ac[it]);
        rb[it] = *(const float4*)(B + (size_t)bk[it] * 256 + col0 + bn_[it]);
    }
#pragma unroll
    for (int it = 0; it < 4; ++it) {
        u16* d = ldsA[0] + ar[it] * 32 + ac[it];
        d[0] = f2bf(ra[it].x); d[1] = f2bf(ra[it].y);
        d[2] = f2bf(ra[it].z); d[3] = f2bf(ra[it].w);
        ldsBt[0][(bn_[it] + 0) * 32 + bk[it]] = f2bf(rb[it].x);
        ldsBt[0][(bn_[it] + 1) * 32 + bk[it]] = f2bf(rb[it].y);
        ldsBt[0][(bn_[it] + 2) * 32 + bk[it]] = f2bf(rb[it].z);
        ldsBt[0][(bn_[it] + 3) * 32 + bk[it]] = f2bf(rb[it].w);
    }
    __syncthreads();

    for (int i = 0; i < 8; ++i) {
        int cur = i & 1;
        // prefetch next k-tile into registers (flies under the WMMAs below)
        if (i < 7) {
            int k0 = (i + 1) * 32;
#pragma unroll
            for (int it = 0; it < 4; ++it) {
                ra[it] = *(const float4*)(A + (size_t)(row0 + ar[it]) * 256 + k0 + ac[it]);
                rb[it] = *(const float4*)(B + (size_t)(k0 + bk[it]) * 256 + col0 + bn_[it]);
            }
        }
        // compute from current LDS buffer
        {
            const u16* LA = ldsA[cur];
            const u16* LB = ldsBt[cur];
            U16F ua;
            ua.h[0] = *(const v8bf*)(LA + (wave * 16 + l16) * 32 + 8 * kg);
            ua.h[1] = *(const v8bf*)(LA + (wave * 16 + l16) * 32 + 16 + 8 * kg);
#pragma unroll
            for (int j = 0; j < 8; ++j) {
                int nloc = j * 16 + l16;
                U16F ub;
                ub.h[0] = *(const v8bf*)(LB + nloc * 32 + 8 * kg);
                ub.h[1] = *(const v8bf*)(LB + nloc * 32 + 16 + 8 * kg);
                acc[j] = __builtin_amdgcn_wmma_f32_16x16x32_bf16(
                    false, ua.v, false, ub.v, (short)0, acc[j], false, false);
            }
        }
        if (i < 7) {
            int nxt = (i + 1) & 1;
            __syncthreads();   // everyone done reading buffer `nxt` (used at i-1)
#pragma unroll
            for (int it = 0; it < 4; ++it) {
                u16* d = ldsA[nxt] + ar[it] * 32 + ac[it];
                d[0] = f2bf(ra[it].x); d[1] = f2bf(ra[it].y);
                d[2] = f2bf(ra[it].z); d[3] = f2bf(ra[it].w);
                ldsBt[nxt][(bn_[it] + 0) * 32 + bk[it]] = f2bf(rb[it].x);
                ldsBt[nxt][(bn_[it] + 1) * 32 + bk[it]] = f2bf(rb[it].y);
                ldsBt[nxt][(bn_[it] + 2) * 32 + bk[it]] = f2bf(rb[it].z);
                ldsBt[nxt][(bn_[it] + 3) * 32 + bk[it]] = f2bf(rb[it].w);
            }
            __syncthreads();
        }
    }

    int mbase = row0 + wave * 16 + 8 * kg;
#pragma unroll
    for (int j = 0; j < 8; ++j) {
        int gn = col0 + j * 16 + l16;
#pragma unroll
        for (int r = 0; r < 8; ++r)
            C[(size_t)(mbase + r) * 256 + gn] = f2bf(acc[j][r]);
    }
}

// ---------------------------------------------------------------------------
// Kernel 5: global average pool of y (bf16) -> p[b,c]
// ---------------------------------------------------------------------------
__global__ void pool_kernel(const u16* __restrict__ ybf, float* __restrict__ p) {
    int bc = blockIdx.x;  // 0..1023
    const u16* Y = ybf + (size_t)bc * 65536u;
    float s = 0.0f;
    for (int ch = threadIdx.x; ch < 8192; ch += 256) {
        int4 v = *(const int4*)(Y + ch * 8);
        const u16* pv = (const u16*)&v;
        for (int j = 0; j < 8; ++j) s += bf2f(pv[j]);
    }
    __shared__ float red[256];
    red[threadIdx.x] = s;
    __syncthreads();
    for (int off = 128; off > 0; off >>= 1) {
        if (threadIdx.x < off) red[threadIdx.x] += red[threadIdx.x + off];
        __syncthreads();
    }
    if (threadIdx.x == 0) p[bc] = red[0] * (1.0f / 65536.0f);
}

// ---------------------------------------------------------------------------
// Kernel 6: channel attention -> fold sigmoid gate into A_b = lc_w * s[b,:]
//           also precompute BN scale/bias. grid=4, block=256.
// ---------------------------------------------------------------------------
__global__ void attn_kernel(const float* __restrict__ p,
                            const float* __restrict__ ca_w1, const float* __restrict__ ca_b1,
                            const float* __restrict__ ca_dw, const float* __restrict__ ca_db,
                            const float* __restrict__ lc_w,
                            const float* __restrict__ bn_g, const float* __restrict__ bn_b,
                            const float* __restrict__ bn_m, const float* __restrict__ bn_v,
                            u16* __restrict__ Ab, float* __restrict__ scaleo,
                            float* __restrict__ biaso) {
    int b = blockIdx.x;
    int c = threadIdx.x;
    __shared__ float pl[256], sl[256];
    pl[c] = p[b * 256 + c];
    __syncthreads();
    float a = ca_b1[c];
    for (int k = 0; k < 256; ++k) a += ca_w1[c * 256 + k] * pl[k];
    a = gelu_f(a);
    a = a * ca_dw[c * 9 + 4] + ca_db[c];
    sl[c] = 1.0f / (1.0f + expf(-a));
    __syncthreads();
    for (int k = 0; k < 256; ++k)
        Ab[((size_t)(b * 256 + c)) * 256 + k] = f2bf(lc_w[c * 256 + k] * sl[k]);
    if (b == 0) {
        float inv = bn_g[c] * rsqrtf(bn_v[c] + 1e-5f);
        scaleo[c] = inv;
        biaso[c] = bn_b[c] - bn_m[c] * inv;
    }
}

// ---------------------------------------------------------------------------
// Kernel 7: out = x + GELU(scale*(A_b @ y) + bias)   (256 x 65536, K=256)
// grid=(2,512,4) block=256. WMMA bf16, double-buffered, fused epilogue.
// ---------------------------------------------------------------------------
__global__ void out_gemm(const u16* __restrict__ Ab, const u16* __restrict__ ybf,
                         const float* __restrict__ x,
                         const float* __restrict__ scaleo, const float* __restrict__ biaso,
                         float* __restrict__ out) {
    int b = blockIdx.z;
    const u16* A = Ab + (size_t)b * 65536u;               // (o, c) lda=256
    const u16* B = ybf + (size_t)b * 256u * 65536u;       // (c, hw) ldb=65536

    __shared__ u16 ldsA[2][128 * 32];
    __shared__ u16 ldsBt[2][128 * 32];

    int tid = threadIdx.x;
    int lane = tid & 31;
    int wave = tid >> 5;
    int row0 = blockIdx.x * 128;
    int col0 = blockIdx.y * 128;
    int kg = lane >> 4;
    int l16 = lane & 15;

    int ar[2], ac[2], bk[2], bn_[2];
#pragma unroll
    for (int it = 0; it < 2; ++it) {
        int idx = tid + it * 256;            // 512 chunks of 8 bf16 per operand
        ar[it] = idx >> 2;  ac[it] = (idx & 3) * 8;    // A: 4 chunks/row
        bk[it] = idx >> 4;  bn_[it] = (idx & 15) * 8;  // B: 16 chunks/k-row
    }

    v8f acc[8];
#pragma unroll
    for (int j = 0; j < 8; ++j) acc[j] = (v8f)(0.0f);

    int4 qa[2], qb[2];
#pragma unroll
    for (int it = 0; it < 2; ++it) {
        qa[it] = *(const int4*)(A + (size_t)(row0 + ar[it]) * 256 + ac[it]);
        qb[it] = *(const int4*)(B + (size_t)bk[it] * 65536u + col0 + bn_[it]);
    }
#pragma unroll
    for (int it = 0; it < 2; ++it) {
        *(int4*)(ldsA[0] + ar[it] * 32 + ac[it]) = qa[it];
        const u16* pv = (const u16*)&qb[it];
#pragma unroll
        for (int j = 0; j < 8; ++j) ldsBt[0][(bn_[it] + j) * 32 + bk[it]] = pv[j];
    }
    __syncthreads();

    for (int i = 0; i < 8; ++i) {
        int cur = i & 1;
        if (i < 7) {
            int k0 = (i + 1) * 32;
#pragma unroll
            for (int it = 0; it < 2; ++it) {
                qa[it] = *(const int4*)(A + (size_t)(row0 + ar[it]) * 256 + k0 + ac[it]);
                qb[it] = *(const int4*)(B + (size_t)(k0 + bk[it]) * 65536u + col0 + bn_[it]);
            }
        }
        {
            const u16* LA = ldsA[cur];
            const u16* LB = ldsBt[cur];
            U16F ua;
            ua.h[0] = *(const v8bf*)(LA + (wave * 16 + l16) * 32 + 8 * kg);
            ua.h[1] = *(const v8bf*)(LA + (wave * 16 + l16) * 32 + 16 + 8 * kg);
#pragma unroll
            for (int j = 0; j < 8; ++j) {
                int nloc = j * 16 + l16;
                U16F ub;
                ub.h[0] = *(const v8bf*)(LB + nloc * 32 + 8 * kg);
                ub.h[1] = *(const v8bf*)(LB + nloc * 32 + 16 + 8 * kg);
                acc[j] = __builtin_amdgcn_wmma_f32_16x16x32_bf16(
                    false, ua.v, false, ub.v, (short)0, acc[j], false, false);
            }
        }
        if (i < 7) {
            int nxt = (i + 1) & 1;
            __syncthreads();
#pragma unroll
            for (int it = 0; it < 2; ++it) {
                *(int4*)(ldsA[nxt] + ar[it] * 32 + ac[it]) = qa[it];
                const u16* pv = (const u16*)&qb[it];
#pragma unroll
                for (int j = 0; j < 8; ++j) ldsBt[nxt][(bn_[it] + j) * 32 + bk[it]] = pv[j];
            }
            __syncthreads();
        }
    }

    int mbase = row0 + wave * 16 + 8 * kg;
#pragma unroll
    for (int j = 0; j < 8; ++j) {
        int gn = col0 + j * 16 + l16;
#pragma unroll
        for (int r = 0; r < 8; ++r) {
            int o = mbase + r;
            float v = scaleo[o] * acc[j][r] + biaso[o];
            float g = gelu_f(v);
            size_t off = ((size_t)(b * 256 + o)) * 65536u + (size_t)gn;
            out[off] = x[off] + g;
        }
    }
}

// ---------------------------------------------------------------------------
extern "C" void kernel_launch(void* const* d_in, const int* in_sizes, int n_in,
                              void* d_out, int out_size, void* d_ws, size_t ws_size,
                              hipStream_t stream) {
    (void)in_sizes; (void)n_in; (void)out_size; (void)ws_size;
    const float* x = (const float*)d_in[0];
    const float* mp[24];
    for (int i = 0; i < 24; ++i) mp[i] = (const float*)d_in[1 + i];
    const float* ca_w1 = (const float*)d_in[25];
    const float* ca_b1 = (const float*)d_in[26];
    const float* ca_dw = (const float*)d_in[27];
    const float* ca_db = (const float*)d_in[28];
    const float* lc_w  = (const float*)d_in[29];
    const float* bn_g  = (const float*)d_in[30];
    const float* bn_b  = (const float*)d_in[31];
    const float* bn_m  = (const float*)d_in[32];
    const float* bn_v  = (const float*)d_in[33];

    float* wsf = (float*)d_ws;
    float* Gh = wsf + OFF_GH;
    float* Gw = wsf + OFF_GW;
    float* Kh = wsf + OFF_KH;
    float* Kw = wsf + OFF_KW;
    float* p  = wsf + OFF_P;
    float* sc = wsf + OFF_SC;
    float* bi = wsf + OFF_BI;
    float* Mh = wsf + OFF_MH;
    float* Mw = wsf + OFF_MW_F;
    u16* Ab  = (u16*)((char*)d_ws + AB_BYTE_OFF);
    u16* ybf = (u16*)((char*)d_ws + YBF_BYTE_OFF);
    float* out = (float*)d_out;

    gates_kernel<<<1, 1024, 0, stream>>>(
        mp[0], mp[1], mp[2], mp[3], mp[4], mp[5],
        mp[6], mp[7], mp[8], mp[9], mp[10], mp[11],
        mp[12], mp[13], mp[14], mp[15], mp[16], mp[17],
        mp[18], mp[19], mp[20], mp[21], mp[22], mp[23],
        Gh, Gw);
    kbuild_kernel<<<dim3(128, 2), 256, 0, stream>>>(Gh, Gw, Kh, Kw);
    mbuild_kernel<<<dim3(128, 2), 256, 0, stream>>>(Kh, Kw, Mh, Mw);
    spectral_gemm<<<dim3(2, 2, 512), 256, 0, stream>>>(Mh, x, ybf, 1);  // HC branch
    spectral_gemm<<<dim3(2, 2, 512), 256, 0, stream>>>(Mw, x, ybf, 0);  // WC branch
    pool_kernel<<<1024, 256, 0, stream>>>(ybf, p);
    attn_kernel<<<4, 256, 0, stream>>>(p, ca_w1, ca_b1, ca_dw, ca_db,
                                       lc_w, bn_g, bn_b, bn_m, bn_v, Ab, sc, bi);
    out_gemm<<<dim3(2, 512, 4), 256, 0, stream>>>(Ab, ybf, x, sc, bi, out);
}
